// DyGCN_1778116460913
// MI455X (gfx1250) — matrix operand order repaired and checked by reading, compiled, not run on hardware
//
#include <hip/hip_runtime.h>
#include <hip/hip_bf16.h>
#include <cstdint>

// ---------------- problem constants ----------------
constexpr int B    = 32;
constexpr int N    = 500;
constexpr int NP   = 512;   // padded N
constexpr int DIN  = 64;
constexpr int DOUT = 64;
constexpr int E    = 16;
constexpr int KCH  = 3;
constexpr int KF   = KCH * DIN;   // 192 (GEMM K of final stage)
constexpr int CE   = E * DOUT;    // 1024 (GEMM N of final stage)

// ---------------- CDNA5 WMMA types ----------------
typedef __attribute__((ext_vector_type(4)))  unsigned int v4u;
typedef __attribute__((ext_vector_type(8)))  unsigned int v8u;
typedef __attribute__((ext_vector_type(8)))  float        v8f;
typedef __attribute__((ext_vector_type(16))) __bf16       v16bf;

static __device__ __forceinline__ unsigned short f32_to_bf16(float f) {
  union { float f; unsigned int u; } v; v.f = f;
  unsigned int u = v.u;
  u += 0x7FFFu + ((u >> 16) & 1u);       // round-to-nearest-even
  return (unsigned short)(u >> 16);
}

static __device__ __forceinline__ v16bf make_frag(v4u q0, v4u q1) {
  v8u u;
  u[0]=q0[0]; u[1]=q0[1]; u[2]=q0[2]; u[3]=q0[3];
  u[4]=q1[0]; u[5]=q1[1]; u[6]=q1[2]; u[7]=q1[3];
  return __builtin_bit_cast(v16bf, u);
}

static __device__ __forceinline__ v8f wmma_bf16(v16bf a, v16bf b, v8f c) {
  // D = A(16x32 bf16) x B(32x16 bf16) + C(16x16 f32)
  return __builtin_amdgcn_wmma_f32_16x16x32_bf16(
      /*neg_a=*/false, a, /*neg_b=*/false, b,
      /*c_mod=*/(short)0, c, /*reuse_a=*/false, /*reuse_b=*/false);
}

// A-operand (16x32, 16-bit): lane m=l&15, hi=l>=16;
//   contiguous K runs: [k+8*hi, +8) and [k+16+8*hi, +8)   -> two b128 loads
static __device__ __forceinline__ v16bf load_afrag(const unsigned short* rowp, int k, int hi) {
  v4u q0 = *reinterpret_cast<const v4u*>(rowp + k + 8 * hi);
  v4u q1 = *reinterpret_cast<const v4u*>(rowp + k + 16 + 8 * hi);
  return make_frag(q0, q1);
}
// B-operand (32x16, 16-bit): lane col=l&15, hi=l>=16;
//   contiguous K runs: [k+8*hi, +8) and [k+16+8*hi, +8)   -> two b128 loads
static __device__ __forceinline__ v16bf load_bfrag(const unsigned short* rowp, int k, int hi) {
  v4u q0 = *reinterpret_cast<const v4u*>(rowp + k + 8 * hi);
  v4u q1 = *reinterpret_cast<const v4u*>(rowp + k + 16 + 8 * hi);
  return make_frag(q0, q1);
}

// ---------------- kernel 1: pack x (bf16) ----------------
// xT0[b][d][n] (B-operand use) and y[b][n][0:64] (A-operand use), pad rows -> 0
__global__ void prep_x_kernel(const float* __restrict__ x,
                              unsigned short* __restrict__ xT0,
                              unsigned short* __restrict__ y) {
  int idx = blockIdx.x * 256 + threadIdx.x;        // B*NP*DIN = 2^20
  if (idx >= B * NP * DIN) return;
  int d = idx & 63;
  int n = (idx >> 6) & (NP - 1);
  int b = idx >> 15;
  float v = (n < N) ? x[((size_t)b * N + n) * DIN + d] : 0.0f;
  unsigned short h = f32_to_bf16(v);
  xT0[((size_t)b * DIN + d) * NP + n] = h;
  y[((size_t)b * NP + n) * KF + d]    = h;
}

// ---------------- kernel 2: pack weights_pool ----------------
// WT[e*64+o][kc*64+i] = wp[e][kc][i][o]   (K contiguous per output column)
__global__ void prep_w_kernel(const float* __restrict__ wp,
                              unsigned short* __restrict__ Wt) {
  int idx = blockIdx.x * 256 + threadIdx.x;        // CE*KF = 196608
  if (idx >= CE * KF) return;
  int c  = idx / KF;
  int ki = idx - c * KF;
  int e  = c >> 6, o = c & 63;
  int kc = ki / DIN, i = ki - kc * DIN;
  float v = wp[(((size_t)e * KCH + kc) * DIN + i) * DOUT + o];
  Wt[(size_t)c * KF + ki] = f32_to_bf16(v);
}

// ---------------- kernel 3: adjacency = softmax(relu(E E^T)) -> bf16 ----------------
// one block per (b, row n); 256 threads cover 512 padded columns
__global__ void adj_softmax_kernel(const float* __restrict__ emb,
                                   unsigned short* __restrict__ A) {
  int n = blockIdx.x;
  int b = blockIdx.y;
  int tid = threadIdx.x;
  unsigned short* Arow = A + ((size_t)b * NP + n) * NP;
  if (n >= N) {                                    // pad rows -> zero (uniform per block)
    for (int m = tid; m < NP; m += 256) Arow[m] = 0;
    return;
  }
  __shared__ float en[E];
  __shared__ float red[256];
  const float* erow = emb + ((size_t)b * N + n) * E;
  if (tid < E) en[tid] = erow[tid];
  __syncthreads();

  float sc0, sc1;
  {
    const float* em = emb + ((size_t)b * N + tid) * E;   // m = tid < 500 always
    float acc = 0.0f;
#pragma unroll
    for (int e = 0; e < E; ++e) acc += en[e] * em[e];
    sc0 = fmaxf(acc, 0.0f);
  }
  {
    int m = tid + 256;
    sc1 = -INFINITY;
    if (m < N) {
      const float* em = emb + ((size_t)b * N + m) * E;
      float acc = 0.0f;
#pragma unroll
      for (int e = 0; e < E; ++e) acc += en[e] * em[e];
      sc1 = fmaxf(acc, 0.0f);
    }
  }
  // block max
  red[tid] = fmaxf(sc0, sc1);
  __syncthreads();
  for (int s = 128; s > 0; s >>= 1) {
    if (tid < s) red[tid] = fmaxf(red[tid], red[tid + s]);
    __syncthreads();
  }
  float mx = red[0];
  __syncthreads();
  float e0 = __expf(sc0 - mx);
  float e1 = (sc1 == -INFINITY) ? 0.0f : __expf(sc1 - mx);
  red[tid] = e0 + e1;
  __syncthreads();
  for (int s = 128; s > 0; s >>= 1) {
    if (tid < s) red[tid] += red[tid + s];
    __syncthreads();
  }
  float inv = 1.0f / red[0];
  Arow[tid]       = f32_to_bf16(e0 * inv);
  Arow[tid + 256] = (tid + 256 < N) ? f32_to_bf16(e1 * inv) : (unsigned short)0;
}

// ---------------- kernel 4/5: X' = A @ X  (WMMA, bf16 in / f32 acc) ----------------
// grid (NP/16, B), 128 threads = 4 waves; wave w owns feature cols [16w,16w+16)
// mode: x_f32 == nullptr -> write acc;  else -> write 2*acc - x (Chebyshev T2)
__global__ void spmm_kernel(const unsigned short* __restrict__ A,   // [B][NP][NP]
                            const unsigned short* __restrict__ Bt,  // [B][DIN][NP]
                            const float* __restrict__ x_f32,        // [B][N][DIN] or null
                            unsigned short* __restrict__ y,         // [B][NP][KF]
                            unsigned short* __restrict__ xT_out,    // [B][DIN][NP] or null
                            int slice) {
  int nb   = blockIdx.x;
  int b    = blockIdx.y;
  int lane = threadIdx.x & 31;
  int wave = threadIdx.x >> 5;
  int hi   = lane >> 4;
  int nn   = lane & 15;
  int col  = wave * 16 + nn;

  const unsigned short* Arow = A  + ((size_t)b * NP + nb * 16 + nn) * NP;
  const unsigned short* Brow = Bt + ((size_t)b * DIN + col) * NP;

  v8f acc = {};
  for (int k = 0; k < NP; k += 32) {
    v16bf af = load_afrag(Arow, k, hi);
    v16bf bf = load_bfrag(Brow, k, hi);
    acc = wmma_bf16(af, bf, acc);
  }
#pragma unroll
  for (int r = 0; r < 8; ++r) {
    int m = r + 8 * hi;                       // C/D layout: VGPR r -> row r (+8 for hi lanes)
    int n = nb * 16 + m;
    float v = acc[r];
    if (x_f32) {
      float xv = (n < N) ? x_f32[((size_t)b * N + n) * DIN + col] : 0.0f;
      v = 2.0f * v - xv;
    }
    unsigned short h = f32_to_bf16(v);
    y[((size_t)b * NP + n) * KF + slice + col] = h;
    if (xT_out) xT_out[((size_t)b * DIN + col) * NP + n] = h;
  }
}

// ---------------- kernel 6: z = y @ WT, contract with station_emb, + bias ----------------
// grid (NP/16, B), 256 threads = 8 waves; wave w owns e in {2w, 2w+1} (8 col tiles)
__global__ void fused_out_kernel(const unsigned short* __restrict__ Ybf, // [B][NP][KF]
                                 const unsigned short* __restrict__ Wt,  // [CE][KF]
                                 const float* __restrict__ st,           // [B][N][E]
                                 const float* __restrict__ bias_pool,    // [E][DOUT]
                                 float* __restrict__ out) {              // [B][N][DOUT]
  __shared__ __align__(32) unsigned short yt[16 * KF];   // 6 KB A-operand tile
  __shared__ float accs[16 * DOUT];                      // 4 KB f32 reduction buffer
  int nb  = blockIdx.x;
  int b   = blockIdx.y;
  int tid = threadIdx.x;

  for (int i = tid; i < 16 * DOUT; i += 256) accs[i] = 0.0f;
  {
    const uint32_t* src = reinterpret_cast<const uint32_t*>(
        Ybf + ((size_t)b * NP + nb * 16) * KF);
    uint32_t* dst = reinterpret_cast<uint32_t*>(yt);
    for (int i = tid; i < (16 * KF) / 2; i += 256) dst[i] = src[i];
  }
  __syncthreads();

  int lane = tid & 31, wave = tid >> 5;
  int hi = lane >> 4, nn = lane & 15;
  const unsigned short* yrow = yt + nn * KF;

  v8f acc[8] = {};
  for (int k = 0; k < KF; k += 32) {
    v16bf af = load_afrag(yrow, k, hi);       // ds_load_b128 x2
#pragma unroll
    for (int t = 0; t < 8; ++t) {
      int c = (wave * 8 + t) * 16 + nn;       // global GEMM column (= e*64 + o)
      v16bf bf = load_bfrag(Wt + (size_t)c * KF, k, hi);
      acc[t] = wmma_bf16(af, bf, acc[t]);
    }
  }

  // station-embedding factors for this wave's two e-values, per owned row
  float se[2][8];
#pragma unroll
  for (int r = 0; r < 8; ++r) {
    int n = nb * 16 + r + 8 * hi;
    if (n < N) {
      const float* sr = st + ((size_t)b * N + n) * E;
      se[0][r] = sr[wave * 2 + 0];
      se[1][r] = sr[wave * 2 + 1];
    } else {
      se[0][r] = 0.0f; se[1][r] = 0.0f;
    }
  }
  // cross-wave E-contraction via LDS float atomics (ds_add_f32)
#pragma unroll
  for (int t = 0; t < 8; ++t) {
    int ct = wave * 8 + t;
    int o  = (ct & 3) * 16 + nn;
    int ei = t >> 2;                          // 0 -> e=2w, 1 -> e=2w+1
#pragma unroll
    for (int r = 0; r < 8; ++r) {
      int m = r + 8 * hi;
      atomicAdd(&accs[m * DOUT + o], se[ei][r] * acc[t][r]);
    }
  }
  __syncthreads();

  // + bias (rank-16) and writeout
  for (int i = tid; i < 16 * DOUT; i += 256) {
    int m = i >> 6, o = i & 63;
    int n = nb * 16 + m;
    if (n < N) {
      const float* sr = st + ((size_t)b * N + n) * E;
      float bsum = 0.0f;
#pragma unroll
      for (int e = 0; e < E; ++e) bsum += sr[e] * bias_pool[e * DOUT + o];
      out[((size_t)b * N + n) * DOUT + o] = accs[i] + bsum;
    }
  }
}

// ---------------- launch ----------------
extern "C" void kernel_launch(void* const* d_in, const int* in_sizes, int n_in,
                              void* d_out, int out_size, void* d_ws, size_t ws_size,
                              hipStream_t stream) {
  const float* x         = (const float*)d_in[0];   // [32,500,64]
  const float* all_emb   = (const float*)d_in[1];   // [32,500,16]
  const float* stat_emb  = (const float*)d_in[2];   // [32,500,16]
  const float* wpool     = (const float*)d_in[3];   // [16,3,64,64]
  const float* bias_pool = (const float*)d_in[4];   // [16,64]
  float* out = (float*)d_out;

  // workspace layout (bf16 stored as ushort); total ~26.4 MB
  unsigned short* A   = (unsigned short*)d_ws;      // [B][NP][NP]   16 MB
  unsigned short* xT0 = A   + (size_t)B * NP * NP;  // [B][DIN][NP]   2 MB
  unsigned short* xT1 = xT0 + (size_t)B * DIN * NP; // [B][DIN][NP]   2 MB
  unsigned short* Y   = xT1 + (size_t)B * DIN * NP; // [B][NP][KF]    6 MB
  unsigned short* Wt  = Y   + (size_t)B * NP * KF;  // [CE][KF]     384 KB

  // 1) pack x -> bf16 (transposed B-operand copy + y k=0 slice)
  prep_x_kernel<<<(B * NP * DIN + 255) / 256, 256, 0, stream>>>(x, xT0, Y);
  // 2) pack weights_pool -> WT
  prep_w_kernel<<<(CE * KF + 255) / 256, 256, 0, stream>>>(wpool, Wt);
  // 3) adjacency softmax -> bf16
  adj_softmax_kernel<<<dim3(NP, B), 256, 0, stream>>>(all_emb, A);
  // 4) x_g1 = A @ x          (write y k=1 slice + transposed copy)
  spmm_kernel<<<dim3(NP / 16, B), 128, 0, stream>>>(A, xT0, nullptr, Y, xT1, DIN);
  // 5) x_g2 = 2*A @ x_g1 - x (write y k=2 slice)
  spmm_kernel<<<dim3(NP / 16, B), 128, 0, stream>>>(A, xT1, x, Y, nullptr, 2 * DIN);
  // 6) out = contract_E( (y @ W) , station_emb ) + station_emb @ bias_pool
  fused_out_kernel<<<dim3(NP / 16, B), 256, 0, stream>>>(Y, Wt, stat_emb, bias_pool, out);
}
